// MicroscaleAttention_12472585027940
// MI455X (gfx1250) — matrix operand order repaired and checked by reading, not compiled, not run on hardware
//
#include <hip/hip_runtime.h>
#include <hip/hip_bf16.h>

typedef __attribute__((ext_vector_type(16))) _Float16 v16h;
typedef __attribute__((ext_vector_type(8)))  float    v8f;

#define B_    4
#define L_    4092
#define H_    8
#define DH    64
#define DM    512      // H_*DH
#define WS_   12
#define STEP_ 6
#define NW    681
#define EPS_  1e-5f

// ---------------- WMMA fragment helpers (wave32, gfx1250 layouts) ----------------

__device__ inline v8f wmma_f16(v16h a, v16h b, v8f c) {
  return __builtin_amdgcn_wmma_f32_16x16x32_f16(
      /*neg_a=*/false, a, /*neg_b=*/false, b,
      /*c_mod=*/(short)0, c, /*reuse_a=*/false, /*reuse_b=*/false);
}

// A-matrix 16x32 (MxK), f16, from row-major LDS: src[m*stride + k]
// lanes 0-15 / 16-31 both hold M=0..15; VGPR v packs K pair: k = 16*half + 2*v
__device__ inline v16h load_A16x32(const _Float16* src, int stride, int k0) {
  int lane = threadIdx.x & 31;
  int hlf = lane >> 4, m = lane & 15;
  v16h a;
#pragma unroll
  for (int v = 0; v < 8; ++v) {
    int kk = k0 + 16 * hlf + 2 * v;
    a[2 * v]     = src[m * stride + kk];
    a[2 * v + 1] = src[m * stride + kk + 1];
  }
  return a;
}

// B-matrix 32x16 (KxN): B[k][n] = M[(n0+n)*64 + k0+k]  (B = M^T tile).
__device__ inline v16h load_B_T(const _Float16* M, int n0, int k0) {
  int lane = threadIdx.x & 31;
  int hlf = lane >> 4, n = lane & 15;
  v16h b;
#pragma unroll
  for (int v = 0; v < 8; ++v) {
    int kk = k0 + 16 * hlf + 2 * v;
    b[2 * v]     = M[(n0 + n) * DH + kk];
    b[2 * v + 1] = M[(n0 + n) * DH + kk + 1];
  }
  return b;
}

// B-matrix 32x16 from row-major source: B[k][n] = M[k*stride + n0+n], rows k >= kmax are 0.
__device__ inline v16h load_B_rm(const _Float16* M, int stride, int n0, int kmax) {
  int lane = threadIdx.x & 31;
  int hlf = lane >> 4, n = lane & 15;
  v16h b;
#pragma unroll
  for (int v = 0; v < 8; ++v) {
    int kk = 16 * hlf + 2 * v;
    b[2 * v]     = (kk     < kmax) ? M[kk * stride + n0 + n]       : (_Float16)0.f;
    b[2 * v + 1] = (kk + 1 < kmax) ? M[(kk + 1) * stride + n0 + n] : (_Float16)0.f;
  }
  return b;
}

// CDNA5 async global->LDS copy, 16 bytes per lane (tracked by ASYNCcnt)
__device__ inline void async_copy16(unsigned int lds_off, const void* gaddr) {
  asm volatile("global_load_async_to_lds_b128 %0, %1, off"
               :: "v"(lds_off), "v"(gaddr) : "memory");
}
__device__ inline void wait_asynccnt0() {
  asm volatile("s_wait_asynccnt 0" ::: "memory");
}

// ---------------- Kernel 1: per-token projections  Qp/Kp/Vp = X @ W^T  (f16 out) ----------------
// grid = (B*H, 32), block = 256 (8 waves; each wave owns one 16-token tile)

__global__ __launch_bounds__(256) void proj_kernel(
    const float* __restrict__ Qin, const float* __restrict__ Kin, const float* __restrict__ Vin,
    const float* __restrict__ Wq,  const float* __restrict__ Wk,  const float* __restrict__ Wv,
    _Float16* __restrict__ Qp, _Float16* __restrict__ Kp, _Float16* __restrict__ Vp)
{
  __shared__ __attribute__((aligned(16))) _Float16 sW[3 * DH * DH];  // 24 KB
  __shared__ __attribute__((aligned(16))) _Float16 sX[8][16 * DH];   // 16 KB: X tile, reused for D

  int tid = threadIdx.x;
  for (int i = tid; i < 3 * DH * DH; i += 256) {
    const float* src = (i < DH * DH) ? Wq : ((i < 2 * DH * DH) ? Wk : Wv);
    sW[i] = (_Float16)src[i & (DH * DH - 1)];
  }
  __syncthreads();

  int wave = tid >> 5, lane = tid & 31;
  int bh = blockIdx.x, b = bh >> 3, h = bh & 7;
  int l0 = (blockIdx.y * 8 + wave) * 16;
  int n = lane & 15, hlf = lane >> 4;

#pragma unroll
  for (int j = 0; j < 3; ++j) {
    const float* X    = (j == 0) ? Qin : ((j == 1) ? Kin : Vin);
    _Float16*    outp = (j == 0) ? Qp  : ((j == 1) ? Kp  : Vp);
    const _Float16* W = &sW[j * DH * DH];

    // load 16x64 f32 input tile -> f16 LDS (zero-pad rows past L)
    for (int i = lane; i < 16 * DH; i += 32) {
      int m = i >> 6, d = i & 63;
      int l = l0 + m;
      float val = (l < L_) ? X[((size_t)(b * L_ + l) * H_ + h) * DH + d] : 0.f;
      sX[wave][i] = (_Float16)val;
      if (j < 2 && l < L_) {
        const float* nx = (j == 0) ? Kin : Vin;
        __builtin_prefetch(&nx[((size_t)(b * L_ + l) * H_ + h) * DH + d], 0, 1);
      }
    }
    __syncthreads();

    v16h a0 = load_A16x32(sX[wave], DH, 0);
    v16h a1 = load_A16x32(sX[wave], DH, 32);
    v8f c[4];
#pragma unroll
    for (int nt = 0; nt < 4; ++nt) {
      v8f acc0 = {};
      acc0 = wmma_f16(a0, load_B_T(W, nt * 16, 0), acc0);
      acc0 = wmma_f16(a1, load_B_T(W, nt * 16, 32), acc0);
      c[nt] = acc0;
    }
    // stage D tile (f16) back into sX (wave-private slice; DS in-order per wave)
#pragma unroll
    for (int nt = 0; nt < 4; ++nt)
#pragma unroll
      for (int r = 0; r < 8; ++r) {
        int m = r + 8 * hlf;
        sX[wave][m * DH + nt * 16 + n] = (_Float16)c[nt][r];
      }
    __syncthreads();

    // coalesced b128 copy to global: 4 chunks of 8 halves per lane
    const uint4* dsrc = reinterpret_cast<const uint4*>(sX[wave]);
#pragma unroll
    for (int it = 0; it < 4; ++it) {
      int i = it * 32 + lane;          // chunk index, 8 halves per chunk
      int m = (i * 8) >> 6;            // row of chunk
      int l = l0 + m;
      if (l < L_) {
        size_t off = (((size_t)(b * H_ + h) * L_ + l) * DH + ((i * 8) & 63)) >> 3;
        reinterpret_cast<uint4*>(outp)[off] = dsrc[i];
      }
    }
    __syncthreads();
  }
}

// ---------------- Kernel 2: windowed attention, one wave per (b,h,window) ----------------
// grid = B*H*NW/4, block = 128 (4 waves)

__global__ __launch_bounds__(128) void attn_kernel(
    const _Float16* __restrict__ Qp, const _Float16* __restrict__ Kp, const _Float16* __restrict__ Vp,
    float* __restrict__ acc, float* __restrict__ attnOut)
{
  __shared__ __attribute__((aligned(16))) _Float16 sQ[4][16 * DH];
  __shared__ __attribute__((aligned(16))) _Float16 sK[4][16 * DH];
  __shared__ __attribute__((aligned(16))) _Float16 sV[4][16 * DH];
  __shared__ __attribute__((aligned(16))) _Float16 sP[4][16 * 32];   // f16 probs, K-padded
  __shared__ __attribute__((aligned(16))) float    sPF[4][16 * 16];  // f32 probs
  __shared__ __attribute__((aligned(16))) float    sO[4][16 * DH];   // f32 output tile

  int wave = threadIdx.x >> 5, lane = threadIdx.x & 31;
  int item = blockIdx.x * 4 + wave;          // grid sized exactly: item < B*H*NW
  int w = item % NW;
  int bh = item / NW;
  int b = bh >> 3, h = bh & 7;
  int l0 = w * STEP_;

  // async DMA of the 12 valid rows (1536 B each = 96 x 16B chunks = 3 iters x 32 lanes)
  size_t base = ((size_t)(b * H_ + h) * L_ + l0) * DH;   // halves; byte offset % 16 == 0
  unsigned int offQ = (unsigned int)(size_t)&sQ[wave][0];
  unsigned int offK = (unsigned int)(size_t)&sK[wave][0];
  unsigned int offV = (unsigned int)(size_t)&sV[wave][0];
#pragma unroll
  for (int it = 0; it < 3; ++it) {
    int i = it * 32 + lane;                  // 16B chunk index
    unsigned int lo = (unsigned int)(i * 16);
    const char* gq = (const char*)(Qp + base) + i * 16;
    const char* gk = (const char*)(Kp + base) + i * 16;
    const char* gv = (const char*)(Vp + base) + i * 16;
    async_copy16(offQ + lo, gq);
    async_copy16(offK + lo, gk);
    async_copy16(offV + lo, gv);
  }
  // zero-pad rows 12..15 (128 dwords per matrix) with plain DS stores
  for (int i = lane; i < 128; i += 32) {
    reinterpret_cast<unsigned int*>(&sQ[wave][WS_ * DH])[i] = 0u;
    reinterpret_cast<unsigned int*>(&sK[wave][WS_ * DH])[i] = 0u;
    reinterpret_cast<unsigned int*>(&sV[wave][WS_ * DH])[i] = 0u;
  }
  wait_asynccnt0();
  __syncthreads();

  // scores = Q @ K^T / 8   (K-dim = 64, two WMMA steps)
  v16h a0 = load_A16x32(sQ[wave], DH, 0);
  v16h a1 = load_A16x32(sQ[wave], DH, 32);
  v8f s = {};
  s = wmma_f16(a0, load_B_T(sK[wave], 0, 0), s);
  s = wmma_f16(a1, load_B_T(sK[wave], 0, 32), s);

  int n = lane & 15, hlf = lane >> 4;
#pragma unroll
  for (int r = 0; r < 8; ++r)
    s[r] = (n < WS_) ? s[r] * 0.125f : -1e30f;   // mask key-cols >= 12

  // row softmax: row m = r + 8*half lives across the 16 lanes of one half
#pragma unroll
  for (int r = 0; r < 8; ++r) {
    float mx = s[r];
    for (int msk = 1; msk < 16; msk <<= 1) mx = fmaxf(mx, __shfl_xor(mx, msk));
    float e = __expf(s[r] - mx);
    float sum = e;
    for (int msk = 1; msk < 16; msk <<= 1) sum += __shfl_xor(sum, msk);
    float p = e / sum;
    int m = r + 8 * hlf;
    sP[wave][m * 32 + n]      = (_Float16)p;
    sP[wave][m * 32 + 16 + n] = (_Float16)0.f;
    sPF[wave][m * 16 + n]     = p;
  }
  __syncthreads();

  // coalesced attn-prob output: 144 contiguous f32 per (b,h,w)
  size_t abase = ((size_t)bh * NW + w) * (WS_ * WS_);
  for (int i = lane; i < WS_ * WS_; i += 32) {
    int q = i / WS_, k = i - q * WS_;
    attnOut[abase + i] = sPF[wave][q * 16 + k];
  }

  // O = P @ V  (K-dim 12 padded to 32), stage f32 O tile in LDS
  v16h ap = load_A16x32(sP[wave], 32, 0);
#pragma unroll
  for (int nt = 0; nt < 4; ++nt) {
    v8f o = {};
    o = wmma_f16(ap, load_B_rm(sV[wave], DH, nt * 16, 16), o);
#pragma unroll
    for (int r = 0; r < 8; ++r) {
      int m = r + 8 * hlf;
      sO[wave][m * DH + nt * 16 + n] = o[r];
    }
  }
  __syncthreads();

  // overlap-add: lane-contiguous atomics over the 12 valid rows
  float* arow = &acc[((size_t)b * L_ + l0) * DM + h * DH];
  for (int i = lane; i < WS_ * DH; i += 32) {
    int m = i >> 6, d = i & 63;
    atomicAdd(&arow[(size_t)m * DM + d], sO[wave][m * DH + d]);
  }
}

// ---------------- Kernel 3: overlap average + residual + LayerNorm ----------------
// one wave per token; lane owns 16 contiguous channels. grid = B*L/8, block = 256.

__global__ __launch_bounds__(256) void norm_kernel(
    const float* __restrict__ acc, const float* __restrict__ Qin,
    const float* __restrict__ gamma, const float* __restrict__ beta,
    float* __restrict__ outZ)
{
  int wave = threadIdx.x >> 5, lane = threadIdx.x & 31;
  int tok = blockIdx.x * 8 + wave;           // grid sized exactly: tok < B*L
  int b = tok / L_, l = tok % L_;
  float inv = (l < STEP_ || l >= L_ - STEP_) ? 1.f : 0.5f;  // 1/cnt
  size_t base = ((size_t)b * L_ + l) * DM;   // residual layout [B,L,H*Dh] == input flat

  float x[16], sum = 0.f, sq = 0.f;
#pragma unroll
  for (int j = 0; j < 16; ++j) {
    int c = lane * 16 + j;
    float v = acc[base + c] * inv + Qin[base + c];
    x[j] = v; sum += v; sq += v * v;
  }
  for (int msk = 1; msk < 32; msk <<= 1) {
    sum += __shfl_xor(sum, msk);
    sq  += __shfl_xor(sq, msk);
  }
  float mu   = sum * (1.f / DM);
  float var  = sq  * (1.f / DM) - mu * mu;
  float rstd = rsqrtf(var + EPS_);
#pragma unroll
  for (int j = 0; j < 16; ++j) {
    int c = lane * 16 + j;
    outZ[base + c] = (x[j] - mu) * rstd * gamma[c] + beta[c];
  }
}

// ---------------- Launch ----------------

extern "C" void kernel_launch(void* const* d_in, const int* in_sizes, int n_in,
                              void* d_out, int out_size, void* d_ws, size_t ws_size,
                              hipStream_t stream) {
  const float* Qin   = (const float*)d_in[0];
  const float* Kin   = (const float*)d_in[1];
  const float* Vin   = (const float*)d_in[2];
  const float* Wq    = (const float*)d_in[3];
  const float* Wk    = (const float*)d_in[4];
  const float* Wv    = (const float*)d_in[5];
  const float* gamma = (const float*)d_in[6];
  const float* beta  = (const float*)d_in[7];

  const size_t nProj = (size_t)B_ * H_ * L_ * DH;     // 8,380,416 halves each
  _Float16* Qp = (_Float16*)d_ws;
  _Float16* Kp = Qp + nProj;
  _Float16* Vp = Kp + nProj;
  float*    acc = (float*)(Vp + nProj);               // [B, L, 512] f32
  const size_t accBytes = (size_t)B_ * L_ * DM * sizeof(float);
  hipMemsetAsync(acc, 0, accBytes, stream);

  float* outZ    = (float*)d_out;                     // [B, L, 512]
  float* outAttn = outZ + (size_t)B_ * L_ * DM;       // [B, H, NW, 12, 12]

  proj_kernel<<<dim3(B_ * H_, 32), 256, 0, stream>>>(Qin, Kin, Vin, Wq, Wk, Wv, Qp, Kp, Vp);
  attn_kernel<<<(B_ * H_ * NW) / 4, 128, 0, stream>>>(Qp, Kp, Vp, acc, outAttn);
  norm_kernel<<<(B_ * L_) / 8, 256, 0, stream>>>(acc, Qin, gamma, beta, outZ);
}
